// WaveFieldAttentionV4D_46024869544527
// MI455X (gfx1250) — matrix operand order repaired and checked by reading, not compiled, not run on hardware
//
#include <hip/hip_runtime.h>
#include <hip/hip_bf16.h>
#include <math.h>

// Problem constants (match reference)
#define B_  4
#define N_  2048
#define D_  1024
#define H_  16
#define G_  4096
#define HD_ 64
#define M_  (B_ * N_)      // 8192 rows for all GEMMs
#define FN  (2 * G_)       // 8192-point FFT
#define FLOG 13

typedef __attribute__((ext_vector_type(16))) _Float16 v16h;
typedef __attribute__((ext_vector_type(8)))  float    v8f;
typedef __attribute__((ext_vector_type(4)))  _Float16 h4;

// ---------------------------------------------------------------------------
// Async global->LDS copy of one 16-byte chunk via the gfx1250 VGLOBAL async
// op (GLOBAL_LOAD_ASYNC_TO_LDS_B128, tracked by ASYNCcnt — ISA 08 §4).
// LDS destination is the 32-bit LDS byte offset (addrspace(3) cast).
// ---------------------------------------------------------------------------
#if defined(__gfx1250__)
__device__ __forceinline__ void async_copy16(const _Float16* gsrc,
                                             _Float16* ldst)
{
  unsigned lds_off = (unsigned)(unsigned long long)(
      (__attribute__((address_space(3))) _Float16*)ldst);
  unsigned long long gaddr = (unsigned long long)gsrc;
  asm volatile("global_load_async_to_lds_b128 %0, %1, off"
               :: "v"(lds_off), "v"(gaddr)
               : "memory");
}
#define ASYNC_WAIT() asm volatile("s_wait_asynccnt 0" ::: "memory")
#else
__device__ __forceinline__ void async_copy16(const _Float16* gsrc,
                                             _Float16* ldst)
{
  *(uint4*)ldst = *(const uint4*)gsrc;
}
#define ASYNC_WAIT() ((void)0)
#endif

// ---------------------------------------------------------------------------
// fp32 -> f16 pre-conversion (vectorized x4); done once per tensor so the
// GEMMs stage pure f16 and the qkv/gate GEMMs share one converted copy of x.
// ---------------------------------------------------------------------------
__global__ __launch_bounds__(256) void k_f32_to_f16(
    const float* __restrict__ in, _Float16* __restrict__ out, int n4)
{
  int i = blockIdx.x * 256 + threadIdx.x;
  if (i < n4) {
    float4 v = ((const float4*)in)[i];
    h4 o = {(_Float16)v.x, (_Float16)v.y, (_Float16)v.z, (_Float16)v.w};
    ((h4*)out)[i] = o;
  }
}

// ---------------------------------------------------------------------------
// WMMA GEMM:  C[M,Nc] = A[M,K] @ W[Nc,K]^T + bias[Nc]
// f16 inputs, fp32 accumulate via v_wmma_f32_16x16x32_f16.
// 256 threads = 8 wave32; block tile 128x64; wave tile 32x32 (2x2 WMMA);
// K stepped by 32 with double-buffered LDS fed by async global->LDS copies
// (3 x b128 per thread per K-step, fully unrolled, no divergence).
// ---------------------------------------------------------------------------
#define TM 128
#define TN 64
#define LP 48   // LDS row stride in halfs: 96B, multiple of 16B, non-pow2 banks

__global__ __launch_bounds__(256) void k_gemm_f16wmma(
    const _Float16* __restrict__ A, const _Float16* __restrict__ W,
    const float* __restrict__ bias, float* __restrict__ C,
    int Mdim, int Ndim, int Kdim)
{
  __shared__ __align__(16) _Float16 As[2][TM][LP];
  __shared__ __align__(16) _Float16 Bs[2][TN][LP];

  const int tid  = threadIdx.x;
  const int lane = tid & 31;
  const int wave = tid >> 5;
  const int wm = wave & 3;          // 4 waves along M
  const int wn = wave >> 2;         // 2 waves along N
  const int half_sel = lane >> 4;   // ISA 7.12.2 lane-half layouts
  const int l16 = lane & 15;
  const int rowBase = blockIdx.y * TM;
  const int colBase = blockIdx.x * TN;

  // Per-thread staging coordinates (compile-time trip counts, no loops):
  // A tile: 128 rows x 4 chunks = 512 chunks -> 2 per thread
  // B tile:  64 rows x 4 chunks = 256 chunks -> 1 per thread
  const int ar0 = tid >> 2,           ac0 = (tid & 3) * 8;
  const int ar1 = (tid + 256) >> 2,   ac1 = ac0;           // same cc pattern
  const int br  = tid >> 2,           bc  = (tid & 3) * 8;

  v8f acc[2][2] = {};

  auto stage = [&](int buf, int k0) {
    async_copy16(&A[(size_t)(rowBase + ar0) * Kdim + k0 + ac0],
                 &As[buf][ar0][ac0]);
    async_copy16(&A[(size_t)(rowBase + ar1) * Kdim + k0 + ac1],
                 &As[buf][ar1][ac1]);
    async_copy16(&W[(size_t)(colBase + br) * Kdim + k0 + bc],
                 &Bs[buf][br][bc]);
  };

  stage(0, 0);
  ASYNC_WAIT();
  __syncthreads();

  const int steps = Kdim / 32;
  for (int it = 0; it < steps; ++it) {
    const int cur = it & 1, nxt = cur ^ 1;
    if (it + 1 < steps) stage(nxt, (it + 1) * 32);   // overlap copy with math

    // A frag: lanes 0-15 hold K 0..7 & 16..23; lanes 16-31 hold 8..15 & 24..31
    v16h afrag[2], bfrag[2];
    for (int rt = 0; rt < 2; ++rt) {
      int row = wm * 32 + rt * 16 + l16;
      int ko  = half_sel ? 8 : 0;
      for (int j = 0; j < 8; ++j) {
        afrag[rt][j]     = As[cur][row][ko + j];
        afrag[rt][8 + j] = As[cur][row][ko + 16 + j];
      }
    }
    // B frag: lane holds one column (row of W); lanes 0-15 K 0..7 & 16..23, 16-31 K 8..15 & 24..31
    for (int ct = 0; ct < 2; ++ct) {
      int n  = wn * 32 + ct * 16 + l16;
      int kb = half_sel * 8;
      for (int j = 0; j < 8; ++j) {
        bfrag[ct][j]     = Bs[cur][n][kb + j];
        bfrag[ct][8 + j] = Bs[cur][n][kb + 16 + j];
      }
    }

    for (int rt = 0; rt < 2; ++rt)
      for (int ct = 0; ct < 2; ++ct)
        acc[rt][ct] = __builtin_amdgcn_wmma_f32_16x16x32_f16(
            false, afrag[rt], false, bfrag[ct], (short)0, acc[rt][ct],
            false, false);

    ASYNC_WAIT();          // nxt buffer landed before next iteration reads it
    __syncthreads();       // and all waves done reading cur before re-staging
  }

  // D layout: VGPR i -> M = i (lanes 0-15) / 8+i (lanes 16-31), N = lane&15
  for (int rt = 0; rt < 2; ++rt)
    for (int ct = 0; ct < 2; ++ct) {
      int gc = colBase + wn * 32 + ct * 16 + l16;
      float bv = bias[gc];
      for (int i = 0; i < 8; ++i) {
        int gr = rowBase + wm * 32 + rt * 16 + half_sel * 8 + i;
        C[(size_t)gr * Ndim + gc] = acc[rt][ct][i] + bv;
      }
    }
}

// ---------------------------------------------------------------------------
// In-LDS radix-2 FFT, 8192 points (64 KB of the 320 KB/WGP LDS), 256 threads.
// sign = -1 forward, +1 inverse (caller scales by 1/FN after inverse).
// ---------------------------------------------------------------------------
__device__ __forceinline__ float2 cmul(float2 a, float2 b) {
  return make_float2(a.x * b.x - a.y * b.y, a.x * b.y + a.y * b.x);
}

__device__ void fft_lds(float2* a, int tid, float sign) {
  for (int i = tid; i < FN; i += 256) {
    unsigned j = __brev((unsigned)i) >> (32 - FLOG);
    if (j > (unsigned)i) { float2 t = a[i]; a[i] = a[(int)j]; a[(int)j] = t; }
  }
  __syncthreads();
  for (int s = 1; s <= FLOG; ++s) {
    int m = 1 << s, mh = m >> 1;
    for (int b = tid; b < FN / 2; b += 256) {
      int blk = b / mh, j = b - blk * mh;
      int i0 = blk * m + j, i1 = i0 + mh;
      float ang = sign * 6.283185307179586f * (float)j / (float)m;
      float sn, cs;
      __sincosf(ang, &sn, &cs);
      float2 u = a[i0];
      float2 t = cmul(make_float2(cs, sn), a[i1]);
      a[i0] = make_float2(u.x + t.x, u.y + t.y);
      a[i1] = make_float2(u.x - t.x, u.y - t.y);
    }
    __syncthreads();
  }
}

// ---------------------------------------------------------------------------
// Build per-head wave kernels: time kernel -> rfft -> dispersion rotation ->
// irfft -> causal truncation -> rfft. Stores kfft[h][0..G]. One block/head.
// ---------------------------------------------------------------------------
__global__ __launch_bounds__(256) void k_build_wave_kernels(
    const float* __restrict__ freq, const float* __restrict__ damp,
    const float* __restrict__ phase, const float* __restrict__ disp,
    float2* __restrict__ kfft)
{
  __shared__ float2 buf[FN];
  __shared__ float  wred[8];
  __shared__ float  norm_s;

  const int h = blockIdx.x, tid = threadIdx.x;
  const float alpha = log1pf(expf(damp[h])) + 0.05f;  // softplus + 0.05
  const float omega = fabsf(freq[h]);
  const float phi   = phase[h];
  const float dsp   = disp[h];

  float part = 0.f;
  for (int t = tid; t < FN; t += 256) {
    float v = 0.f;
    if (t < G_) v = expf(-alpha * (float)t) * cosf(omega * (float)t + phi);
    buf[t] = make_float2(v, 0.f);
    part += fabsf(v);
  }
  for (int off = 16; off; off >>= 1) part += __shfl_down(part, off, 32);
  if ((tid & 31) == 0) wred[tid >> 5] = part;
  __syncthreads();
  if (tid == 0) {
    float s = 0.f;
    for (int i = 0; i < 8; ++i) s += wred[i];
    norm_s = 1.f / fmaxf(s, 1e-8f);
  }
  __syncthreads();
  const float nrm = norm_s;
  for (int t = tid; t < G_; t += 256) buf[t].x *= nrm;
  __syncthreads();

  fft_lds(buf, tid, -1.f);

  for (int i = tid; i <= G_; i += 256) {
    float fn = (float)i / (float)G_;
    float ph = dsp * fn * fn * 6.283185307179586f;
    float sn, cs;
    __sincosf(ph, &sn, &cs);
    buf[i] = cmul(buf[i], make_float2(cs, sn));
  }
  __syncthreads();
  for (int i = G_ + 1 + tid; i < FN; i += 256) {   // Hermitian mirror
    float2 v = buf[FN - i];
    buf[i] = make_float2(v.x, -v.y);
  }
  __syncthreads();

  fft_lds(buf, tid, +1.f);

  const float inv = 1.f / (float)FN;
  for (int t = tid; t < FN; t += 256) {
    float v = (t < G_) ? buf[t].x * inv : 0.f;     // causal truncation
    buf[t] = make_float2(v, 0.f);
  }
  __syncthreads();

  fft_lds(buf, tid, -1.f);
  for (int i = tid; i <= G_; i += 256) kfft[h * (G_ + 1) + i] = buf[i];
}

// ---------------------------------------------------------------------------
// FFT convolution of one (b,h,d) line with kfft[h]. B*H*HD = 4096 blocks.
// ---------------------------------------------------------------------------
__global__ __launch_bounds__(256) void k_fft_conv(
    const float* __restrict__ field, const float2* __restrict__ kfft,
    float* __restrict__ out)
{
  __shared__ float2 buf[FN];
  const int bid = blockIdx.x;
  const int d = bid % HD_;
  const int h = (bid / HD_) % H_;
  const int b = bid / (HD_ * H_);
  const int tid = threadIdx.x;
  const size_t base = (((size_t)b * H_ + h) * G_) * HD_ + d;

  for (int t = tid; t < FN; t += 256) {
    float v = (t < G_) ? field[base + (size_t)t * HD_] : 0.f;
    buf[t] = make_float2(v, 0.f);
  }
  __syncthreads();

  fft_lds(buf, tid, -1.f);

  const float2* kh = kfft + h * (G_ + 1);
  for (int i = tid; i < FN; i += 256) {
    float2 kb;
    if (i <= G_) kb = kh[i];
    else { float2 t = kh[FN - i]; kb = make_float2(t.x, -t.y); }  // Hermitian
    buf[i] = cmul(buf[i], kb);
  }
  __syncthreads();

  fft_lds(buf, tid, +1.f);

  const float inv = 1.f / (float)FN;
  for (int t = tid; t < G_; t += 256)
    out[base + (size_t)t * HD_] = buf[t].x * inv;
}

// ---------------------------------------------------------------------------
// Softmax over rows of the H x H coupling matrix.
// ---------------------------------------------------------------------------
__global__ void k_coupling_softmax(const float* __restrict__ fc,
                                   float* __restrict__ csm)
{
  int i = threadIdx.x;
  if (i < H_) {
    float mx = -1e30f;
    for (int j = 0; j < H_; ++j) mx = fmaxf(mx, fc[i * H_ + j]);
    float e[H_], s = 0.f;
    for (int j = 0; j < H_; ++j) { e[j] = expf(fc[i * H_ + j] - mx); s += e[j]; }
    for (int j = 0; j < H_; ++j) csm[i * H_ + j] = e[j] / s;
  }
}

// ---------------------------------------------------------------------------
// Bilinear position helper (matches reference clipping exactly).
// ---------------------------------------------------------------------------
__device__ __forceinline__ void pos_weights(int n, int& lo, int& hi,
                                            float& wlo, float& whi)
{
  const float stride = (float)(G_ - 1) / (float)(N_ - 1);
  float pos = fminf(fmaxf((float)n * stride, 0.f), (float)(G_ - 2));
  int l = (int)pos;
  l = min(max(l, 0), G_ - 2);
  float frac = fminf(fmaxf(pos - (float)l, 0.f), 1.f);
  lo = l; hi = l + 1; wlo = 1.f - frac; whi = frac;
}

// ---------------------------------------------------------------------------
// Deposit: field[b,h,lo/hi,d] += v * ||k||_h * w. One block per (b,n).
// ---------------------------------------------------------------------------
__global__ __launch_bounds__(256) void k_deposit(
    const float* __restrict__ qkv, float* __restrict__ field)
{
  __shared__ float ksq[H_];
  __shared__ float kmag[H_];
  const int bn = blockIdx.x;
  const int b = bn / N_, n = bn % N_;
  const int tid = threadIdx.x;
  if (tid < H_) ksq[tid] = 0.f;
  __syncthreads();

  const size_t rowb = (size_t)bn * (3 * D_);
  for (int c = tid; c < D_; c += 256) {
    float kv = qkv[rowb + D_ + c];
    atomicAdd(&ksq[c >> 6], kv * kv);
  }
  __syncthreads();
  if (tid < H_) kmag[tid] = sqrtf(ksq[tid]);
  __syncthreads();

  int lo, hi; float wlo, whi;
  pos_weights(n, lo, hi, wlo, whi);

  for (int c = tid; c < D_; c += 256) {
    int h = c >> 6, d = c & 63;
    float v   = qkv[rowb + 2 * D_ + c];
    float dep = v * kmag[h];
    size_t fb = ((size_t)b * H_ + h) * G_;
    atomicAdd(&field[(fb + (size_t)lo) * HD_ + d], dep * wlo);
    atomicAdd(&field[(fb + (size_t)hi) * HD_ + d], dep * whi);
  }
}

// ---------------------------------------------------------------------------
// Apply H x H coupling at every (b,g) slice: fout[i,:] = sum_j csm[i,j] fin[j,:]
// ---------------------------------------------------------------------------
__global__ __launch_bounds__(256) void k_couple(
    const float* __restrict__ fin, const float* __restrict__ csm,
    float* __restrict__ fout)
{
  __shared__ float sh[H_][HD_];
  __shared__ float cs[H_ * H_];
  const int bg = blockIdx.x;
  const int b = bg / G_, g = bg % G_;
  const int tid = threadIdx.x;
  if (tid < H_ * H_) cs[tid] = csm[tid];
  for (int i = tid; i < H_ * HD_; i += 256) {
    int j = i >> 6, d = i & 63;
    sh[j][d] = fin[(((size_t)b * H_ + j) * G_ + g) * HD_ + d];
  }
  __syncthreads();
  for (int i = tid; i < H_ * HD_; i += 256) {
    int hi_ = i >> 6, d = i & 63;
    float s = 0.f;
    for (int j = 0; j < H_; ++j) s += cs[hi_ * H_ + j] * sh[j][d];
    fout[(((size_t)b * H_ + hi_) * G_ + g) * HD_ + d] = s;
  }
}

// ---------------------------------------------------------------------------
// Gather field at bilinear positions, apply sigmoid gate, emit f16 (B,N,D)
// ready for the output-projection WMMA GEMM.
// ---------------------------------------------------------------------------
__global__ __launch_bounds__(256) void k_gather_gate(
    const float* __restrict__ field3, const float* __restrict__ gatel,
    _Float16* __restrict__ y)
{
  const int bn = blockIdx.x;
  const int b = bn / N_, n = bn % N_;
  const int tid = threadIdx.x;
  int lo, hi; float wlo, whi;
  pos_weights(n, lo, hi, wlo, whi);
  for (int c = tid; c < D_; c += 256) {
    int h = c >> 6, d = c & 63;
    size_t fb = ((size_t)b * H_ + h) * G_;
    float glo = field3[(fb + (size_t)lo) * HD_ + d];
    float ghi = field3[(fb + (size_t)hi) * HD_ + d];
    float gath = glo * wlo + ghi * whi;
    float gl = gatel[(size_t)bn * D_ + c];
    float gate = 1.f / (1.f + expf(-gl));
    y[(size_t)bn * D_ + c] = (_Float16)(gath * gate);
  }
}

// ---------------------------------------------------------------------------
// Orchestration. Workspace (256B aligned), ~313 MB total.
// ---------------------------------------------------------------------------
extern "C" void kernel_launch(void* const* d_in, const int* in_sizes, int n_in,
                              void* d_out, int out_size, void* d_ws,
                              size_t ws_size, hipStream_t stream)
{
  (void)in_sizes; (void)n_in; (void)out_size; (void)ws_size;

  const float* x      = (const float*)d_in[0];
  const float* W_qkv  = (const float*)d_in[1];
  const float* b_qkv  = (const float*)d_in[2];
  const float* W_out  = (const float*)d_in[3];
  const float* b_out  = (const float*)d_in[4];
  const float* W_gate = (const float*)d_in[5];
  const float* b_gate = (const float*)d_in[6];
  const float* wfreq  = (const float*)d_in[7];
  const float* wdamp  = (const float*)d_in[8];
  const float* wphase = (const float*)d_in[9];
  const float* wdisp  = (const float*)d_in[10];
  const float* fcoup  = (const float*)d_in[11];
  float* out = (float*)d_out;

  char* ws = (char*)d_ws;
  size_t off = 0;
  auto alloc = [&](size_t bytes) -> void* {
    void* p = ws + off;
    off += (bytes + 255) & ~(size_t)255;
    return p;
  };
  const size_t fieldBytes = (size_t)B_ * H_ * G_ * HD_ * sizeof(float);
  float*     qkv    = (float*)alloc((size_t)M_ * 3 * D_ * sizeof(float));
  float*     gatel  = (float*)alloc((size_t)M_ * D_ * sizeof(float));
  float*     field  = (float*)alloc(fieldBytes);
  float*     field2 = (float*)alloc(fieldBytes);
  float2*    kfft   = (float2*)alloc((size_t)H_ * (G_ + 1) * sizeof(float2));
  float*     csm    = (float*)alloc((size_t)H_ * H_ * sizeof(float));
  _Float16*  xh     = (_Float16*)alloc((size_t)M_ * D_ * sizeof(_Float16));
  _Float16*  Wqkvh  = (_Float16*)alloc((size_t)3 * D_ * D_ * sizeof(_Float16));
  _Float16*  Wgateh = (_Float16*)alloc((size_t)D_ * D_ * sizeof(_Float16));
  _Float16*  Wouth  = (_Float16*)alloc((size_t)D_ * D_ * sizeof(_Float16));
  _Float16*  yh     = (_Float16*)alloc((size_t)M_ * D_ * sizeof(_Float16));

  // Stage 0: f32->f16 pre-conversion for all GEMM operands
  k_f32_to_f16<<<(M_ * D_ / 4 + 255) / 256, 256, 0, stream>>>(x, xh, M_ * D_ / 4);
  k_f32_to_f16<<<(3 * D_ * D_ / 4 + 255) / 256, 256, 0, stream>>>(W_qkv, Wqkvh, 3 * D_ * D_ / 4);
  k_f32_to_f16<<<(D_ * D_ / 4 + 255) / 256, 256, 0, stream>>>(W_gate, Wgateh, D_ * D_ / 4);
  k_f32_to_f16<<<(D_ * D_ / 4 + 255) / 256, 256, 0, stream>>>(W_out, Wouth, D_ * D_ / 4);

  // Stage 0b: per-head kernel spectra + coupling softmax (tiny, independent)
  k_build_wave_kernels<<<H_, 256, 0, stream>>>(wfreq, wdamp, wphase, wdisp, kfft);
  k_coupling_softmax<<<1, 32, 0, stream>>>(fcoup, csm);

  // Stage 1: QKV and gate GEMMs (WMMA f16, fp32 accumulate, async-LDS staged)
  k_gemm_f16wmma<<<dim3(3 * D_ / TN, M_ / TM), 256, 0, stream>>>(
      xh, Wqkvh, b_qkv, qkv, M_, 3 * D_, D_);
  k_gemm_f16wmma<<<dim3(D_ / TN, M_ / TM), 256, 0, stream>>>(
      xh, Wgateh, b_gate, gatel, M_, D_, D_);

  // Stage 2: zero field, scatter deposit
  hipMemsetAsync(field, 0, fieldBytes, stream);
  k_deposit<<<B_ * N_, 256, 0, stream>>>(qkv, field);

  // Stage 3: FFT convolution per (b,h,d) line
  k_fft_conv<<<B_ * H_ * HD_, 256, 0, stream>>>(field, kfft, field2);

  // Stage 4: head coupling (field reused as output buffer)
  k_couple<<<B_ * G_, 256, 0, stream>>>(field2, csm, field);

  // Stage 5: gather + gate -> f16
  k_gather_gate<<<B_ * N_, 256, 0, stream>>>(field, gatel, yh);

  // Stage 6: output projection GEMM (WMMA) -> d_out
  k_gemm_f16wmma<<<dim3(D_ / TN, M_ / TM), 256, 0, stream>>>(
      yh, Wouth, b_out, out, M_, D_, D_);
}
